// Attention_3246995275789
// MI455X (gfx1250) — matrix operand order repaired and checked by reading, not compiled, not run on hardware
//
#include <hip/hip_runtime.h>
#include <hip/hip_bf16.h>

typedef __attribute__((ext_vector_type(16))) _Float16 v16h;
typedef __attribute__((ext_vector_type(8)))  _Float16 v8h;
typedef __attribute__((ext_vector_type(4)))  _Float16 v4h;
typedef __attribute__((ext_vector_type(8)))  float    v8f;
typedef __attribute__((ext_vector_type(4)))  int      v4i;

#define GLOBAL_AS __attribute__((address_space(1)))
#define LDS_AS    __attribute__((address_space(3)))

union AV { v16h v; v8h h[2]; };

constexpr int B_   = 4;
constexpr int L_   = 2048;
constexpr int DIN  = 1024;
constexpr int H_   = 16;
constexpr int DH   = 64;
constexpr int MTOT = B_ * L_;            // 8192 tokens
constexpr int NQK  = 2 * H_ * DH;        // 2048 qk columns
constexpr int NTOT = NQK + H_ * DH;      // 3072 total proj columns

static __device__ __forceinline__ v8f wmma_f16(v16h a, v16h b, v8f c) {
  // (neg_a, A, neg_b, B, c_mod, C, reuse_a, reuse_b)
  return __builtin_amdgcn_wmma_f32_16x16x32_f16(false, a, false, b, (short)0, c,
                                                false, false);
}

static __device__ __forceinline__ v8f zero8() {
  v8f z;
#pragma unroll
  for (int i = 0; i < 8; ++i) z[i] = 0.0f;
  return z;
}

// ---- CDNA5 async copy: GLOBAL -> LDS, 16B per lane, tracked by ASYNCcnt ----
static __device__ __forceinline__ void async_b128(const _Float16* g,
                                                  _Float16* l) {
#if __has_builtin(__builtin_amdgcn_global_load_async_to_lds_b128)
  __builtin_amdgcn_global_load_async_to_lds_b128(
      (GLOBAL_AS v4i*)(uintptr_t)g, (LDS_AS v4i*)l, 0, 0);
#else
  uint32_t loff = (uint32_t)(uintptr_t)(LDS_AS void*)l;
  asm volatile("global_load_async_to_lds_b128 %0, %1, off" ::"v"(loff),
               "v"((uint64_t)(uintptr_t)g)
               : "memory");
#endif
}

static __device__ __forceinline__ void wait_async_le3() {
#if __has_builtin(__builtin_amdgcn_s_wait_asynccnt)
  __builtin_amdgcn_s_wait_asynccnt(3);
#else
  asm volatile("s_wait_asynccnt 3" ::: "memory");
#endif
}

// ---------------------------------------------------------------- cvt kernels
__global__ void cvt_x_kernel(const float* __restrict__ x,
                             _Float16* __restrict__ xh) {
  int i = (blockIdx.x * blockDim.x + threadIdx.x) * 4;
  float4 f = *(const float4*)(x + i);
  v4h o;
  o[0] = (_Float16)f.x; o[1] = (_Float16)f.y;
  o[2] = (_Float16)f.z; o[3] = (_Float16)f.w;
  *(v4h*)(xh + i) = o;
}

// wt[c][k] = W[k][c] for the fused [Wqk | Wv] weight, cast to f16.
__global__ void cvt_w_kernel(const float* __restrict__ Wqk,
                             const float* __restrict__ Wv,
                             _Float16* __restrict__ wt) {
  int idx = blockIdx.x * blockDim.x + threadIdx.x;  // over DIN*NTOT, c fastest
  int k = idx / NTOT;
  int c = idx - k * NTOT;
  float v = (c < NQK) ? Wqk[(size_t)k * NQK + c]
                      : Wv[(size_t)k * (H_ * DH) + (c - NQK)];
  wt[(size_t)c * DIN + k] = (_Float16)v;
}

// ------------------------------------------------------------ projection GEMM
__global__ __launch_bounds__(128) void gemm_proj_kernel(
    const _Float16* __restrict__ xh,   // [8192,1024]
    const _Float16* __restrict__ wt,   // [3072,1024] (transposed weights)
    const float* __restrict__ bqk, const float* __restrict__ bv,
    _Float16* __restrict__ Qm, _Float16* __restrict__ Km,
    _Float16* __restrict__ Vt) {
  const int lane = threadIdx.x & 31;
  const int wid  = threadIdx.x >> 5;
  const int ln   = lane & 15;
  const int hi   = lane >> 4;
  const int wg   = blockIdx.x * 4 + wid;
  const int nt   = wg % (NTOT / 64);
  const int mt   = wg / (NTOT / 64);

  v8f acc[4];
#pragma unroll
  for (int t = 0; t < 4; ++t) acc[t] = zero8();

  const _Float16* ap = xh + (size_t)(mt * 16 + ln) * DIN;
#pragma unroll 4
  for (int kk = 0; kk < DIN; kk += 32) {
    AV a;
    a.h[0] = *(const v8h*)(ap + kk + hi * 16);
    a.h[1] = *(const v8h*)(ap + kk + 8 + hi * 16);
#pragma unroll
    for (int t = 0; t < 4; ++t) {
      const _Float16* bp =
          wt + (size_t)(nt * 64 + t * 16 + ln) * DIN + kk + hi * 16;
      v16h b = *(const v16h*)bp;
      acc[t] = wmma_f16(a.v, b, acc[t]);
    }
  }

  // Epilogue: (acc + bias) / 32, scatter as f16 into attention layouts.
  const float scale = 1.0f / 32.0f;
#pragma unroll
  for (int t = 0; t < 4; ++t) {
    const int gc = nt * 64 + t * 16 + ln;  // global output column
#pragma unroll
    for (int r = 0; r < 8; ++r) {
      const int gm = mt * 16 + r + hi * 8;  // global token row
      const int bb = gm >> 11;              // / L_
      const int l  = gm & (L_ - 1);
      float bias = (gc < NQK) ? bqk[gc] : bv[gc - NQK];
      _Float16 hv = (_Float16)((acc[t][r] + bias) * scale);
      if (gc < NQK) {
        const int head = gc >> 7;      // /128
        const int w    = gc & 127;
        if (w < DH)
          Qm[(((size_t)(bb * H_ + head) * L_ + l) * DH) + w] = hv;
        else
          Km[(((size_t)(bb * H_ + head) * L_ + l) * DH) + (w - DH)] = hv;
      } else {
        const int vc   = gc - NQK;
        const int head = vc >> 6;
        const int d    = vc & 63;
        Vt[(((size_t)(bb * H_ + head) * DH + d) * L_) + l] = hv;
      }
    }
  }
}

// ------------------------------------------------------------ flash attention
// 4 waves / block share one (b,h); KV tiles are staged into LDS with async
// copies (double buffered), each wave owns 16 query rows.
__global__ __launch_bounds__(128) void attn_kernel(
    const _Float16* __restrict__ Qm, const _Float16* __restrict__ Km,
    const _Float16* __restrict__ Vt, float* __restrict__ out) {
  __shared__ alignas(64) _Float16 kbuf[2][32 * 64];  // 2 x 4KB K tiles
  __shared__ alignas(64) _Float16 vbuf[2][64 * 32];  // 2 x 4KB V^T tiles
  __shared__ alignas(32) _Float16 plds[4 * 16 * 32]; // P tiles, 1KB/wave

  const int tid  = threadIdx.x;
  const int lane = tid & 31;
  const int wid  = tid >> 5;
  const int ln   = lane & 15;
  const int hi   = lane >> 4;
  const int wg   = blockIdx.x * 4 + wid;
  const int qb   = wg & (L_ / 16 - 1);  // query block, 0..127
  const int bh   = wg >> 7;             // fused (b,h); same for whole block

  const _Float16* Qb = Qm + ((size_t)bh * L_ + qb * 16) * DH;
  const _Float16* Kb = Km + (size_t)bh * L_ * DH;
  const _Float16* Vb = Vt + (size_t)bh * DH * L_;
  _Float16* pl = plds + wid * (16 * 32);

  // Stage one 32-key KV block into LDS: K tile is 4KB contiguous (rows are
  // d-major), V^T tile is 64 rows x 64B. 3 async b128 ops per thread.
  auto prefetch = [&](int s, int j0) {
    const _Float16* kg = Kb + (size_t)j0 * DH;
    async_b128(kg + tid * 8, kbuf[s] + tid * 8);
    async_b128(kg + (tid + 128) * 8, kbuf[s] + (tid + 128) * 8);
    const int vr = tid >> 1, vo = (tid & 1) * 8;
    async_b128(Vb + (size_t)vr * L_ + j0 + vo, vbuf[s] + vr * 32 + vo);
  };

  prefetch(0, 0);

  // Resident Q A-fragments (16x64 = two 16x32 chunks).
  AV qa[2];
  {
    const _Float16* qp = Qb + ln * DH;
#pragma unroll
    for (int c = 0; c < 2; ++c) {
      qa[c].h[0] = *(const v8h*)(qp + c * 32 + hi * 16);
      qa[c].h[1] = *(const v8h*)(qp + c * 32 + 8 + hi * 16);
    }
  }

  v8f o[4];
  float rm[8], rs[8];
#pragma unroll
  for (int t = 0; t < 4; ++t) o[t] = zero8();
#pragma unroll
  for (int r = 0; r < 8; ++r) { rm[r] = -__builtin_inff(); rs[r] = 0.0f; }

  const float sscale = 1.0f / 64.0f;  // 1/dim attention scale

#pragma unroll 1
  for (int j0 = 0; j0 < L_; j0 += 32) {
    const int s  = (j0 >> 5) & 1;
    const int jn = (j0 + 32 < L_) ? (j0 + 32) : 0;  // dummy wrap on last iter
    prefetch(s ^ 1, jn);
    wait_async_le3();   // newest 3 outstanding -> current buffer complete
    __syncthreads();    // other waves' stages visible

    const _Float16* kb = kbuf[s];
    const _Float16* vb = vbuf[s];

    // ---- S = Q K^T for 16 q-rows x 32 keys (two 16x16 tiles), from LDS
    const _Float16* kp0 = kb + ln * DH + hi * 16;
    const _Float16* kp1 = kp0 + 16 * DH;
    v8f s0 = zero8(), s1 = zero8();
    s0 = wmma_f16(qa[0].v, *(const v16h*)(kp0), s0);
    s0 = wmma_f16(qa[1].v, *(const v16h*)(kp0 + 32), s0);
    s1 = wmma_f16(qa[0].v, *(const v16h*)(kp1), s1);
    s1 = wmma_f16(qa[1].v, *(const v16h*)(kp1 + 32), s1);

    // ---- online softmax (row lives across 16 lanes of each half-wave)
    float p0[8], p1[8], bm[8], ps[8];
#pragma unroll
    for (int r = 0; r < 8; ++r) {
      p0[r] = s0[r] * sscale;
      p1[r] = s1[r] * sscale;
      bm[r] = fmaxf(p0[r], p1[r]);
    }
#pragma unroll
    for (int mask = 1; mask < 16; mask <<= 1)
#pragma unroll
      for (int r = 0; r < 8; ++r)
        bm[r] = fmaxf(bm[r], __shfl_xor(bm[r], mask, 32));
#pragma unroll
    for (int r = 0; r < 8; ++r) {
      float nm   = fmaxf(rm[r], bm[r]);
      float corr = __expf(rm[r] - nm);
      rm[r] = nm;
      float e0 = __expf(p0[r] - nm);
      float e1 = __expf(p1[r] - nm);
      p0[r] = e0; p1[r] = e1;
      ps[r] = e0 + e1;
      rs[r] *= corr;
#pragma unroll
      for (int t = 0; t < 4; ++t) o[t][r] *= corr;
    }
#pragma unroll
    for (int mask = 1; mask < 16; mask <<= 1)
#pragma unroll
      for (int r = 0; r < 8; ++r) ps[r] += __shfl_xor(ps[r], mask, 32);
#pragma unroll
    for (int r = 0; r < 8; ++r) rs[r] += ps[r];

    // ---- C-layout -> A-layout transpose of P via per-wave LDS tile.
    //      DS ops from one wave are in-order; no barrier needed.
#pragma unroll
    for (int r = 0; r < 8; ++r) {
      const int row = r + hi * 8;
      pl[row * 32 + ln]      = (_Float16)p0[r];
      pl[row * 32 + 16 + ln] = (_Float16)p1[r];
    }
    AV pa;
    pa.h[0] = *(const v8h*)(pl + ln * 32 + hi * 16);
    pa.h[1] = *(const v8h*)(pl + ln * 32 + 8 + hi * 16);

    // ---- O += P @ V from the LDS V^T tile (row stride 32)
#pragma unroll
    for (int t = 0; t < 4; ++t) {
      v16h bv16 = *(const v16h*)(vb + (t * 16 + ln) * 32 + hi * 16);
      o[t] = wmma_f16(pa.v, bv16, o[t]);
    }

    __syncthreads();  // nobody still reads buffer s when it is re-staged
  }

  // ---- normalize and write f32 output [b, l, h*d]
  const int b = bh >> 4, h = bh & 15;
#pragma unroll
  for (int r = 0; r < 8; ++r) {
    float inv = 1.0f / rs[r];
    const int row = qb * 16 + r + hi * 8;
    float* op = out + ((size_t)(b * L_ + row)) * (H_ * DH) + h * DH + ln;
#pragma unroll
    for (int t = 0; t < 4; ++t) op[t * 16] = o[t][r] * inv;
  }
}

// ----------------------------------------------------------------- launcher
extern "C" void kernel_launch(void* const* d_in, const int* in_sizes, int n_in,
                              void* d_out, int out_size, void* d_ws,
                              size_t ws_size, hipStream_t stream) {
  const float* x   = (const float*)d_in[0];
  const float* Wqk = (const float*)d_in[1];
  const float* bqk = (const float*)d_in[2];
  const float* Wv  = (const float*)d_in[3];
  const float* bv  = (const float*)d_in[4];
  float* out = (float*)d_out;

  char* ws = (char*)d_ws;
  const size_t xh_bytes = (size_t)MTOT * DIN * 2;        // 16 MB
  const size_t wt_bytes = (size_t)NTOT * DIN * 2;        //  6 MB
  const size_t qkv_one  = (size_t)B_ * H_ * L_ * DH * 2; // 16 MB each
  _Float16* xh = (_Float16*)ws;
  _Float16* wt = (_Float16*)(ws + xh_bytes);
  _Float16* Qm = (_Float16*)(ws + xh_bytes + wt_bytes);
  _Float16* Km = (_Float16*)((char*)Qm + qkv_one);
  _Float16* Vt = (_Float16*)((char*)Km + qkv_one);

  // 1) f32 -> f16 conversions (x and transposed fused weights)
  cvt_x_kernel<<<(MTOT * DIN / 4) / 256, 256, 0, stream>>>(x, xh);
  cvt_w_kernel<<<(DIN * NTOT) / 256, 256, 0, stream>>>(Wqk, Wv, wt);

  // 2) fused QKV projection GEMM (WMMA), scatter to attention layouts
  const int gemm_waves = (MTOT / 16) * (NTOT / 64);  // 512 * 48
  gemm_proj_kernel<<<gemm_waves / 4, 128, 0, stream>>>(xh, wt, bqk, bv, Qm, Km,
                                                       Vt);

  // 3) flash attention (WMMA + async LDS staging)
  const int attn_waves = B_ * H_ * (L_ / 16);  // 8192
  attn_kernel<<<attn_waves / 4, 128, 0, stream>>>(Qm, Km, Vt, out);
}
